// WindowAttention_29944511988016
// MI455X (gfx1250) — matrix operand order repaired and checked by reading, not compiled, not run on hardware
//
#include <hip/hip_runtime.h>
#include <hip/hip_bf16.h>

typedef __attribute__((ext_vector_type(16))) __bf16 v16bf;
typedef __attribute__((ext_vector_type(8)))  float  v8f;
typedef unsigned short u16;
typedef unsigned int   u32;

// round-half-up bf16 (1 add + 1 shift), pair-pack via v_perm_b32 (2 adds + 1 perm)
__device__ __forceinline__ u16 f2bf(float f) {
  return (u16)((__float_as_uint(f) + 0x8000u) >> 16);
}
__device__ __forceinline__ u32 pack2bf(float lo, float hi) {
  u32 a = __float_as_uint(lo) + 0x8000u;
  u32 b = __float_as_uint(hi) + 0x8000u;
  return __builtin_amdgcn_perm(b, a, 0x07060302u);  // {b.hi16, a.hi16}
}
__device__ __forceinline__ float bf2f(u16 h) {
  return __uint_as_float(((u32)h) << 16);
}

// A fragment (16x32 bf16, M x K): lane m = lm, K-half = lh*8; e0..7 -> k0+e, e8..15 -> k0+16+e-8
__device__ __forceinline__ v16bf ldfragA(const u16* base, int ld, int mt, int kt, int lm, int lh) {
  const u16* p = base + (mt * 16 + lm) * ld + kt * 32 + lh * 8;
  union { v16bf v; uint4 q[2]; } u;
  u.q[0] = *(const uint4*)p;
  u.q[1] = *(const uint4*)(p + 16);
  return u.v;
}
// B fragment (32x16 bf16, K x N) from row-major [N][K] storage (i.e. W for x@W^T):
// lane n = lm, K-half = lh*8; e0..7 -> k0+e, e8..15 -> k0+16+e-8
__device__ __forceinline__ v16bf ldfragB(const u16* base, int ld, int nt, int kt, int lm, int lh) {
  const u16* p = base + (nt * 16 + lm) * ld + kt * 32 + lh * 8;
  union { v16bf v; uint4 q[2]; } u;
  u.q[0] = *(const uint4*)p;
  u.q[1] = *(const uint4*)(p + 16);
  return u.v;
}
__device__ __forceinline__ v8f wmma_bf16(v16bf a, v16bf b, v8f c) {
  return __builtin_amdgcn_wmma_f32_16x16x32_bf16(false, a, false, b, (short)0, c, false, false);
}

// ---------------- prep kernel 1: CPB MLP -> rpb[H][64][64], logit scales ----------------
__global__ void prep_cpb(const float* __restrict__ rel_table, const float* __restrict__ w1,
                         const float* __restrict__ b1, const float* __restrict__ w2,
                         const int* __restrict__ rel_index, const float* __restrict__ logit_scale,
                         float* __restrict__ rpb, float* __restrict__ scales) {
  __shared__ float bt[225 * 4];
  int tid = threadIdx.x;
  if (tid < 4) scales[tid] = __expf(fminf(logit_scale[tid], logf(100.0f)));
  for (int r = tid; r < 225; r += 256) {
    float t0 = rel_table[r * 2 + 0], t1 = rel_table[r * 2 + 1];
    float a0 = 0.f, a1 = 0.f, a2 = 0.f, a3 = 0.f;
    for (int s = 0; s < 512; ++s) {
      float hv = fmaxf(t0 * w1[s * 2 + 0] + t1 * w1[s * 2 + 1] + b1[s], 0.0f);
      a0 += hv * w2[0 * 512 + s];
      a1 += hv * w2[1 * 512 + s];
      a2 += hv * w2[2 * 512 + s];
      a3 += hv * w2[3 * 512 + s];
    }
    bt[r * 4 + 0] = a0; bt[r * 4 + 1] = a1; bt[r * 4 + 2] = a2; bt[r * 4 + 3] = a3;
  }
  __syncthreads();
  for (int i = tid; i < 4 * 64 * 64; i += 256) {
    int h = i >> 12;
    int ij = i & 4095;
    float v = bt[rel_index[ij] * 4 + h];
    rpb[i] = 16.0f / (1.0f + __expf(-v));
  }
}

// ---------------- prep kernel 2: fp32 weights -> bf16 workspace ----------------
__global__ void prep_weights(const float* __restrict__ Wqkv, const float* __restrict__ Wproj,
                             u16* __restrict__ wq, u16* __restrict__ wp) {
  int i = blockIdx.x * 256 + threadIdx.x;
  if (i < 384 * 128) wq[i] = f2bf(Wqkv[i]);
  if (i < 128 * 128) wp[i] = f2bf(Wproj[i]);
}

// ---------------- fused window-attention kernel: one block per window ----------------
__global__ __launch_bounds__(128, 1) void swin_attn(
    const float* __restrict__ x,
    const float* __restrict__ qbias, const float* __restrict__ vbias,
    const float* __restrict__ pbias,
    const u16* __restrict__ Wq, const u16* __restrict__ Wp,
    const float* __restrict__ rpb, const float* __restrict__ scales,
    float* __restrict__ out) {
  __shared__ __align__(16) u16 sm[32768];   // 64 KB
  u16* xb  = sm;          // [64][128] x (bf16)            -> reused as P heads 0,1
  u16* qb  = sm + 8192;   // [64][128] q                   -> reused as P heads 2,3
  u16* kb  = sm + 16384;  // [64][128] k                   -> reused as O (bf16)
  u16* vbT = sm + 24576;  // [128][64] v transposed (persists)
  u16* ob  = sm + 16384;

  const int tid  = threadIdx.x;
  const int lane = tid & 31;
  const int lm = lane & 15;
  const int lh = lane >> 4;
  // wave id / head id as provably-uniform SGPR value -> scalar branches + scalar addressing
  const int h = __builtin_amdgcn_readfirstlane(tid >> 5);
  u16* Pb = sm + h * 4096;              // [64][64] per-head P

  const float* xg = x + (size_t)blockIdx.x * 8192;
  __builtin_prefetch((const void*)Wq, 0, 3);

  // Phase 1: stage x into LDS as bf16
  for (int i = tid; i < 2048; i += 128) {
    float4 f = ((const float4*)xg)[i];
    *(uint2*)(xb + i * 4) = make_uint2(pack2bf(f.x, f.y), pack2bf(f.z, f.w));
  }
  __syncthreads();

  // Phase 2: qkv = x @ Wqkv^T + [q_bias, 0, v_bias]   (M=64, N=384, K=128)
  for (int ct = h; ct < 24; ct += 4) {   // ct uniform: ct<8 -> q, ct<16 -> k, else v
    int c = ct * 16 + lm;
    float bias = (ct < 8) ? qbias[c] : (ct < 16 ? 0.0f : vbias[c - 256]);
    v16bf Bf[4];
    #pragma unroll
    for (int kt = 0; kt < 4; ++kt) Bf[kt] = ldfragB(Wq, 128, ct, kt, lm, lh);
    #pragma unroll
    for (int mt = 0; mt < 4; ++mt) {
      v8f acc;
      #pragma unroll
      for (int i = 0; i < 8; ++i) acc[i] = bias;
      #pragma unroll
      for (int kt = 0; kt < 4; ++kt)
        acc = wmma_bf16(ldfragA(xb, 128, mt, kt, lm, lh), Bf[kt], acc);
      if (ct < 8) {                       // q: row-major, pair-pack then lo/hi b16 stores
        #pragma unroll
        for (int j = 0; j < 4; ++j) {
          u32 w = pack2bf(acc[2 * j], acc[2 * j + 1]);
          int m = mt * 16 + 2 * j + lh * 8;
          qb[m * 128 + c]       = (u16)w;
          qb[(m + 1) * 128 + c] = (u16)(w >> 16);
        }
      } else if (ct < 16) {               // k: row-major
        int cc = c - 128;
        #pragma unroll
        for (int j = 0; j < 4; ++j) {
          u32 w = pack2bf(acc[2 * j], acc[2 * j + 1]);
          int m = mt * 16 + 2 * j + lh * 8;
          kb[m * 128 + cc]       = (u16)w;
          kb[(m + 1) * 128 + cc] = (u16)(w >> 16);
        }
      } else {                            // v: transposed, 8 contiguous rows -> one b128
        int cc = c - 256;
        union { uint4 q; u32 w[4]; } u;
        #pragma unroll
        for (int j = 0; j < 4; ++j) u.w[j] = pack2bf(acc[2 * j], acc[2 * j + 1]);
        *(uint4*)(vbT + cc * 64 + mt * 16 + lh * 8) = u.q;
      }
    }
  }
  __syncthreads();

  // Phase 3: cosine-normalize q (fold per-head scale) and k, vectorized LDS access
  for (int t = tid; t < 512; t += 128) {
    int isK = t & 1;
    int hh  = (t >> 1) & 3;
    int m   = t >> 3;
    u16* p = (isK ? kb : qb) + m * 128 + hh * 32;
    union { uint4 q[4]; u32 w[16]; } u;
    #pragma unroll
    for (int j = 0; j < 4; ++j) u.q[j] = ((uint4*)p)[j];
    float f[32]; float ss = 0.f;
    #pragma unroll
    for (int j = 0; j < 16; ++j) {
      float lo = bf2f((u16)(u.w[j] & 0xFFFFu));
      float hi = bf2f((u16)(u.w[j] >> 16));
      f[2 * j] = lo; f[2 * j + 1] = hi;
      ss += lo * lo + hi * hi;
    }
    float sc = 1.0f / fmaxf(sqrtf(ss), 1e-12f);
    if (!isK) sc *= scales[hh];
    #pragma unroll
    for (int j = 0; j < 16; ++j) u.w[j] = pack2bf(f[2 * j] * sc, f[2 * j + 1] * sc);
    #pragma unroll
    for (int j = 0; j < 4; ++j) ((uint4*)p)[j] = u.q[j];
  }
  __syncthreads();

  // Phase 4: S = qn @ kn^T  (per head; K=32 -> single WMMA per 16x16 tile), in registers
  v8f S[4][4];
  #pragma unroll
  for (int mt = 0; mt < 4; ++mt) {
    v16bf a = ldfragA(qb + h * 32, 128, mt, 0, lm, lh);
    #pragma unroll
    for (int nt = 0; nt < 4; ++nt) {
      v16bf bb = ldfragB(kb + h * 32, 128, nt, 0, lm, lh);
      v8f z;
      #pragma unroll
      for (int i = 0; i < 8; ++i) z[i] = 0.f;
      S[mt][nt] = wmma_bf16(a, bb, z);
    }
  }
  __syncthreads();   // all q/k reads done; Pb may now overlay xb/qb

  // Phase 5: + rpb, row softmax via 16-lane shuffles, write P (bf16) to LDS
  const float* rpb_h = rpb + h * 4096;
  #pragma unroll
  for (int mt = 0; mt < 4; ++mt) {
    float sv[4][8];
    #pragma unroll
    for (int nt = 0; nt < 4; ++nt)
      #pragma unroll
      for (int i = 0; i < 8; ++i) {
        int m = mt * 16 + i + lh * 8;
        int n = nt * 16 + lm;
        sv[nt][i] = S[mt][nt][i] + rpb_h[m * 64 + n];
      }
    #pragma unroll
    for (int i = 0; i < 8; ++i) {
      float mx = fmaxf(fmaxf(sv[0][i], sv[1][i]), fmaxf(sv[2][i], sv[3][i]));
      mx = fmaxf(mx, __shfl_xor(mx, 1));
      mx = fmaxf(mx, __shfl_xor(mx, 2));
      mx = fmaxf(mx, __shfl_xor(mx, 4));
      mx = fmaxf(mx, __shfl_xor(mx, 8));
      float e0 = __expf(sv[0][i] - mx), e1 = __expf(sv[1][i] - mx);
      float e2 = __expf(sv[2][i] - mx), e3 = __expf(sv[3][i] - mx);
      float sum = e0 + e1 + e2 + e3;
      sum += __shfl_xor(sum, 1);
      sum += __shfl_xor(sum, 2);
      sum += __shfl_xor(sum, 4);
      sum += __shfl_xor(sum, 8);
      float inv = 1.0f / sum;
      int m = mt * 16 + i + lh * 8;
      Pb[m * 64 +  0 + lm] = f2bf(e0 * inv);
      Pb[m * 64 + 16 + lm] = f2bf(e1 * inv);
      Pb[m * 64 + 32 + lm] = f2bf(e2 * inv);
      Pb[m * 64 + 48 + lm] = f2bf(e3 * inv);
    }
  }
  __syncthreads();

  // Phase 6: O = P @ v   (A = Pb [64][64], B columns from transposed vbT)
  #pragma unroll
  for (int mt = 0; mt < 4; ++mt) {
    #pragma unroll
    for (int vt = 0; vt < 2; ++vt) {
      v8f acc;
      #pragma unroll
      for (int i = 0; i < 8; ++i) acc[i] = 0.f;
      #pragma unroll
      for (int kt = 0; kt < 2; ++kt) {
        v16bf a  = ldfragA(Pb, 64, mt, kt, lm, lh);
        v16bf bb = ldfragB(vbT + (h * 32 + vt * 16) * 64, 64, 0, kt, lm, lh);
        acc = wmma_bf16(a, bb, acc);
      }
      #pragma unroll
      for (int j = 0; j < 4; ++j) {
        u32 w = pack2bf(acc[2 * j], acc[2 * j + 1]);
        int m = mt * 16 + 2 * j + lh * 8;
        int cc = h * 32 + vt * 16 + lm;
        ob[m * 128 + cc]       = (u16)w;
        ob[(m + 1) * 128 + cc] = (u16)(w >> 16);
      }
    }
  }
  __syncthreads();

  // Phase 7: out = O @ proj_w^T + proj_b, stream to HBM
  float* og = out + (size_t)blockIdx.x * 8192;
  for (int nt = h; nt < 8; nt += 4) {
    int c = nt * 16 + lm;
    float bias = pbias[c];
    v16bf Bf[4];
    #pragma unroll
    for (int kt = 0; kt < 4; ++kt) Bf[kt] = ldfragB(Wp, 128, nt, kt, lm, lh);
    #pragma unroll
    for (int mt = 0; mt < 4; ++mt) {
      v8f acc;
      #pragma unroll
      for (int i = 0; i < 8; ++i) acc[i] = bias;
      #pragma unroll
      for (int kt = 0; kt < 4; ++kt)
        acc = wmma_bf16(ldfragA(ob, 128, mt, kt, lm, lh), Bf[kt], acc);
      #pragma unroll
      for (int i = 0; i < 8; ++i) {
        int m = mt * 16 + i + lh * 8;
        og[m * 128 + c] = acc[i];
      }
    }
  }
}

extern "C" void kernel_launch(void* const* d_in, const int* in_sizes, int n_in,
                              void* d_out, int out_size, void* d_ws, size_t ws_size,
                              hipStream_t stream) {
  const float* x           = (const float*)d_in[0];
  const float* W_qkv       = (const float*)d_in[1];
  const float* q_bias      = (const float*)d_in[2];
  const float* v_bias      = (const float*)d_in[3];
  const float* logit_scale = (const float*)d_in[4];
  const float* cpb_w1      = (const float*)d_in[5];
  const float* cpb_b1      = (const float*)d_in[6];
  const float* cpb_w2      = (const float*)d_in[7];
  const float* proj_w      = (const float*)d_in[8];
  const float* proj_b      = (const float*)d_in[9];
  const float* rel_table   = (const float*)d_in[10];
  const int*   rel_index   = (const int*)d_in[11];
  float* out = (float*)d_out;

  char* ws = (char*)d_ws;
  float* rpb      = (float*)ws;                       // 65536 B: [4][64][64]
  float* scales   = (float*)(ws + 65536);             // 16 B (padded)
  u16*   Wqkv_bf  = (u16*)(ws + 66048);               // 98304 B: [384][128]
  u16*   Wproj_bf = (u16*)(ws + 66048 + 98304);       // 32768 B: [128][128]

  prep_cpb<<<1, 256, 0, stream>>>(rel_table, cpb_w1, cpb_b1, cpb_w2,
                                  rel_index, logit_scale, rpb, scales);
  prep_weights<<<192, 256, 0, stream>>>(W_qkv, proj_w, Wqkv_bf, Wproj_bf);
  swin_attn<<<8192, 128, 0, stream>>>(x, q_bias, v_bias, proj_b,
                                      Wqkv_bf, Wproj_bf, rpb, scales, out);
}
